// BinaryTreeLSTM_58557584113864
// MI455X (gfx1250) — matrix operand order repaired and checked by reading, not compiled, not run on hardware
//
#include <hip/hip_runtime.h>
#include <hip/hip_bf16.h>
#include <math.h>

// ---------------------------------------------------------------------------
// BinaryTreeLSTM forward for MI455X (gfx1250), wave32 + WMMA bf16 + TDM.
// B=32, L=32, D=512, H=512.
// ---------------------------------------------------------------------------

typedef __bf16 bf16_t;
typedef __attribute__((ext_vector_type(16))) __bf16 v16bf;
typedef __attribute__((ext_vector_type(8)))  __bf16 v8bf;
typedef __attribute__((ext_vector_type(8)))  float  v8f;
typedef unsigned int u32x4 __attribute__((ext_vector_type(4)));
typedef int          i32x8 __attribute__((ext_vector_type(8)));
typedef int          i32x4 __attribute__((ext_vector_type(4)));

static constexpr int Bc = 32;
static constexpr int Lc = 32;
static constexpr int Dc = 512;
static constexpr int Hc = 512;
static constexpr int P  = Lc - 1;        // 31 positions per step
static constexpr int ROWS = Bc * P;      // 992 rows in the comp GEMM
static constexpr int NW   = 2 * Hc;      // 1024  (word-proj N)
static constexpr int KW   = Dc;          // 512   (word-proj K)
static constexpr int NC   = 5 * Hc;      // 2560  (comp N)
static constexpr int KC   = 2 * Hc;      // 1024  (comp K)

// GEMM blocking
static constexpr int BM   = 64;          // block rows
static constexpr int BN   = 256;         // block cols
static constexpr int KBLK = 128;         // K chunk staged per TDM transfer
static constexpr int RSB  = KBLK * 2 + 16; // LDS row stride: 256B + 16B TDM pad
static constexpr int BUFB = BN * RSB;      // 69632 bytes per buffer

__device__ __forceinline__ float sigmoidf_(float x) {
    return 1.0f / (1.0f + __expf(-x));
}

// ---------------------------------------------------------------------------
// TDM: DMA a 2D bf16 tile (KBLK elems contiguous per row, BN rows, row
// stride = K elements) from global into LDS at byte offset lds_off.
// D# built per CDNA5 ISA 8.3/8.4:
//   group0: count=1 | lds_addr | global_addr | type=2
//   group1: data_size=2B, pad_enable, pad_interval=64 DWORDs, pad_amount=4
//           DWORDs (-> 272B LDS row stride, de-conflicts 64-bank LDS),
//           tensor dims/strides, tile dims (tile_dim2=0 -> 2D).
// Groups 2/3 unused for 2D tiles (zero). This toolchain exposes the 6-arg
// builtin: (u32x4, i32x8, i32x4, i32x4, i32x8, i32 cpol).
// ---------------------------------------------------------------------------
__device__ __forceinline__ void tdm_load_tile(unsigned int lds_off,
                                              const bf16_t* gsrc, int K) {
    const unsigned long long ga = (unsigned long long)(uintptr_t)gsrc;
    u32x4 g0;
    g0[0] = 1u;                                        // count=1, user D#
    g0[1] = lds_off;                                   // lds_addr (bytes)
    g0[2] = (unsigned int)(ga & 0xFFFFFFFFu);          // global_addr[31:0]
    g0[3] = (unsigned int)((ga >> 32) & 0x01FFFFFFu)   // global_addr[56:32]
          | (2u << 30);                                // type=2 ("image")
    i32x8 g1;
    g1[0] = (1 << 16)        // data_size: 2 bytes
          | (1 << 20)        // pad_enable
          | (5 << 22)        // pad_interval: 64 DWORDs (=256B row)
          | (3 << 25);       // pad_amount: 4 DWORDs (=16B)
    g1[1] = (K & 0xFFFF) << 16;                        // tensor_dim0[15:0]
    g1[2] = ((K >> 16) & 0xFFFF)                       // tensor_dim0[31:16]
          | ((BN & 0xFFFF) << 16);                     // tensor_dim1[15:0]
    g1[3] = 0 | ((KBLK & 0xFFFF) << 16);               // dim1 hi | tile_dim0
    g1[4] = BN & 0xFFFF;                               // tile_dim1 (tile_dim2=0)
    g1[5] = K;                                         // tensor_dim0_stride lo
    g1[6] = 0;                                         // stride hi | dim1_stride lo
    g1[7] = 0;
    const i32x4 z4 = {0, 0, 0, 0};
    const i32x8 z8 = {0, 0, 0, 0, 0, 0, 0, 0};
    __builtin_amdgcn_tensor_load_to_lds(g0, g1, z4, z4, z8, 0);
}

// ---------------------------------------------------------------------------
// Block GEMM core: C(BM x BN) tile, 8 waves as 4(M) x 2(N), wave tile 16x128
// (8 fp32 accumulators). Weight tile double-buffered in LDS via TDM; A
// fragments straight from global (L2-resident, 2x16B per lane per K-step).
//
// A fragment (ISA 16-bit A 16x32): lane t, g=t>>4, r=t&15:
//   elems 0..7  = K (g*8..g*8+7), elems 8..15 = K (16+g*8..16+g*8+7).
// B fragment from LDS tile (row n: RSB-byte stride): lane t: row r of the
//   16-col subtile, 8 K at byte col ks*64 + g*16 and 8 K at +32 (same k order as A).
// ---------------------------------------------------------------------------
__device__ __forceinline__ void gemm_core_wmma(const bf16_t* __restrict__ A,
                                               const bf16_t* __restrict__ Wt,
                                               int K, int Mtot,
                                               int blockM, int blockN,
                                               char* smem, v8f acc[8]) {
    const int w  = threadIdx.x >> 5;
    const int wm = w & 3;
    const int wn = w >> 2;
    const int t  = threadIdx.x & 31;
    const int g  = t >> 4;
    const int r  = t & 15;

    int arow = blockM + wm * 16 + r;
    if (arow >= Mtot) arow = Mtot - 1;                 // ragged-edge clamp
    const bf16_t* ap = A + (size_t)arow * K + g * 8;

    const bf16_t* wtile = Wt + (size_t)blockN * K;     // tile row start
    const unsigned int lds_base = (unsigned int)(uintptr_t)smem;

    const int nchunks = K / KBLK;
    if (w == 0) tdm_load_tile(lds_base, wtile, K);

    for (int kc = 0; kc < nchunks; ++kc) {
        const int cur = kc & 1;
        __syncthreads();   // readers of buf[cur^1] (prev iter) are done
        if (w == 0 && kc + 1 < nchunks)
            tdm_load_tile(lds_base + (unsigned)(cur ^ 1) * BUFB,
                          wtile + (size_t)(kc + 1) * KBLK, K);
        if (kc + 1 < nchunks) __builtin_amdgcn_s_wait_tensorcnt(1);
        else                  __builtin_amdgcn_s_wait_tensorcnt(0);
        __syncthreads();   // chunk kc landed in LDS, visible to all waves

        const char* buf = smem + cur * BUFB;
        const bf16_t* apc = ap + kc * KBLK;
        #pragma unroll
        for (int ks = 0; ks < KBLK / 32; ++ks) {
            __builtin_prefetch(apc + ks * 32 + KBLK, 0, 3);
            union { v16bf v; v8bf h[2]; } a;
            a.h[0] = *(const v8bf*)(apc + ks * 32);
            a.h[1] = *(const v8bf*)(apc + ks * 32 + 16);
            const int colb = ks * 64 + g * 16;
            #pragma unroll
            for (int j = 0; j < 8; ++j) {
                const int nl = wn * 128 + j * 16 + r;
                union { v16bf v; v8bf h[2]; } b;
                b.h[0] = *(const v8bf*)(buf + nl * RSB + colb);
                b.h[1] = *(const v8bf*)(buf + nl * RSB + colb + 32);
                acc[j] = __builtin_amdgcn_wmma_f32_16x16x32_bf16(
                    false, a.v, false, b.v, (short)0, acc[j], false, false);
            }
        }
    }
}

// ---------------------------------------------------------------------------
// GEMM 1: hc = inp_bf16 @ W_word + b_word ; split into h0 / c0 buffers.
// Grid: (NW/BN, M/BM), block 256, dynamic LDS = 2*BUFB.
// ---------------------------------------------------------------------------
__global__ void __launch_bounds__(256)
gemm_word_kernel(const bf16_t* __restrict__ A, const bf16_t* __restrict__ Wt,
                 const float* __restrict__ bias,
                 float* __restrict__ hbuf, float* __restrict__ cbuf) {
    extern __shared__ char smem[];
    const int blockM = blockIdx.y * BM;
    const int blockN = blockIdx.x * BN;
    v8f acc[8] = {};
    gemm_core_wmma(A, Wt, KW, Bc * Lc, blockM, blockN, smem, acc);

    const int w = threadIdx.x >> 5, wm = w & 3, wn = w >> 2;
    const int t = threadIdx.x & 31, col = t & 15, g = t >> 4;
    const int tileM = blockM + wm * 16;
    for (int j = 0; j < 8; ++j) {
        const int gn = blockN + wn * 128 + j * 16 + col;
        const float bv = bias[gn];
        #pragma unroll
        for (int v = 0; v < 8; ++v) {
            const int gr = tileM + g * 8 + v;
            if (gr < Bc * Lc) {
                const float val = acc[j][v] + bv;
                if (gn < Hc) hbuf[(size_t)gr * Hc + gn]        = val;
                else         cbuf[(size_t)gr * Hc + (gn - Hc)] = val;
            }
        }
    }
}

// ---------------------------------------------------------------------------
// GEMM 2: gates = Acat @ W_comp + b_comp. Grid: (NC/BN, ceil(ROWS/BM)).
// ---------------------------------------------------------------------------
__global__ void __launch_bounds__(256)
gemm_comp_kernel(const bf16_t* __restrict__ A, const bf16_t* __restrict__ Wt,
                 const float* __restrict__ bias, float* __restrict__ gates) {
    extern __shared__ char smem[];
    const int blockM = blockIdx.y * BM;
    const int blockN = blockIdx.x * BN;
    v8f acc[8] = {};
    gemm_core_wmma(A, Wt, KC, ROWS, blockM, blockN, smem, acc);

    const int w = threadIdx.x >> 5, wm = w & 3, wn = w >> 2;
    const int t = threadIdx.x & 31, col = t & 15, g = t >> 4;
    const int tileM = blockM + wm * 16;
    for (int j = 0; j < 8; ++j) {
        const int gn = blockN + wn * 128 + j * 16 + col;
        const float bv = bias[gn];
        #pragma unroll
        for (int v = 0; v < 8; ++v) {
            const int gr = tileM + g * 8 + v;
            if (gr < ROWS) gates[(size_t)gr * NC + gn] = acc[j][v] + bv;
        }
    }
}

// ---------------------------------------------------------------------------
// Prep kernels: fp32 -> bf16 conversion / transpose.
// ---------------------------------------------------------------------------
__global__ void cvt_bf16_kernel(const float* __restrict__ x, bf16_t* __restrict__ y, int n) {
    int i = blockIdx.x * blockDim.x + threadIdx.x;
    if (i < n) y[i] = (bf16_t)x[i];
}

// W is [K][N] row-major fp32; Wt is [N][K] row-major bf16.
__global__ void cvt_transpose_kernel(const float* __restrict__ W, bf16_t* __restrict__ Wt,
                                     int K, int N) {
    int i = blockIdx.x * blockDim.x + threadIdx.x;
    if (i >= N * K) return;
    const int n = i / K;
    const int k = i - n * K;
    Wt[i] = (bf16_t)W[(size_t)k * N + n];
}

// Build Acat bf16 (ROWS x KC): row (b,p): [h[b,p,:], h[b,p+1,:]].
__global__ void pack_kernel(const float* __restrict__ h, bf16_t* __restrict__ Acat) {
    int i = blockIdx.x * blockDim.x + threadIdx.x;
    if (i >= ROWS * KC) return;
    const int row = i >> 10;           // KC = 1024
    const int kc  = i & (KC - 1);
    const int b = row / P;
    const int p = row - b * P;
    float v;
    if (kc < Hc) v = h[((size_t)(b * Lc + p)) * Hc + kc];
    else         v = h[((size_t)(b * Lc + p + 1)) * Hc + (kc - Hc)];
    Acat[i] = (bf16_t)v;
}

// Copy leaves h0 into nodes[:, 0:L, :].
__global__ void copy_h0_kernel(const float* __restrict__ h, float* __restrict__ nodes) {
    int i = blockIdx.x * blockDim.x + threadIdx.x;
    if (i >= Bc * Lc * Hc) return;
    const int b = i / (Lc * Hc);
    const int rest = i - b * (Lc * Hc);
    const int l = rest / Hc;
    const int hh = rest - l * Hc;
    nodes[((size_t)(b * (2 * Lc - 1) + l)) * Hc + hh] = h[i];
}

// ---------------------------------------------------------------------------
// Cell: LSTM gates -> nh, nc and per-row logit = dot(nh, comp_query).
// One block per row (992 rows), 256 threads.
// ---------------------------------------------------------------------------
__global__ void __launch_bounds__(256)
cell_kernel(const float* __restrict__ gates, const float* __restrict__ c_old,
            const float* __restrict__ cq,
            float* __restrict__ nh, float* __restrict__ nc,
            float* __restrict__ logits) {
    const int row = blockIdx.x;      // b*31 + p
    const int b = row / P;
    const int p = row - b * P;
    const float* g = gates + (size_t)row * NC;
    const float* cl_base = c_old + ((size_t)(b * Lc + p)) * Hc;
    const float* cr_base = c_old + ((size_t)(b * Lc + p + 1)) * Hc;

    __shared__ float red[256];
    float acc = 0.0f;
    for (int h = threadIdx.x; h < Hc; h += 256) {
        const float ig = g[h];
        const float fl = g[Hc + h];
        const float fr = g[2 * Hc + h];
        const float u  = g[3 * Hc + h];
        const float o  = g[4 * Hc + h];
        const float c  = cl_base[h] * sigmoidf_(fl + 1.0f)
                       + cr_base[h] * sigmoidf_(fr + 1.0f)
                       + tanhf(u) * sigmoidf_(ig);
        const float hh = sigmoidf_(o) * tanhf(c);
        nh[(size_t)row * Hc + h] = hh;
        nc[(size_t)row * Hc + h] = c;
        acc += hh * cq[h];
    }
    red[threadIdx.x] = acc;
    __syncthreads();
    for (int s = 128; s > 0; s >>= 1) {
        if (threadIdx.x < s) red[threadIdx.x] += red[threadIdx.x + s];
        __syncthreads();
    }
    if (threadIdx.x == 0) logits[row] = red[0];
}

// ---------------------------------------------------------------------------
// Per-batch argmax + state recombination. One block per batch element.
// ---------------------------------------------------------------------------
__global__ void __launch_bounds__(256)
update_kernel(const float* __restrict__ h_old, const float* __restrict__ c_old,
              const float* __restrict__ nh, const float* __restrict__ nc,
              const float* __restrict__ logits, const int* __restrict__ length,
              float* __restrict__ h_new, float* __restrict__ c_new,
              float* __restrict__ nodes, float* __restrict__ out_hf,
              float* __restrict__ out_cf, int step) {
    const int b = blockIdx.x;
    __shared__ int   s_sel;
    __shared__ float s_done;
    __shared__ int   s_last;
    if (threadIdx.x == 0) {
        const int len   = length[b];
        const int limit = len - 1 - step;       // candidates: p < limit
        float best = -3.4e38f;
        int bi = 0;
        for (int p = 0; p < P; ++p) {
            const float v = (p < limit) ? logits[b * P + p] : -1.0e9f;
            if (v > best) { best = v; bi = p; }  // first max, like jnp.argmax
        }
        s_sel  = bi;
        s_done = (step + 1 < len) ? 1.0f : 0.0f;
        s_last = (step == P - 1) ? 1 : 0;
    }
    __syncthreads();
    const int   ksel    = s_sel;
    const float done    = s_done;
    const int   is_last = s_last;

    float* node_out = nodes + ((size_t)(b * (2 * Lc - 1) + Lc + step)) * Hc;

    for (int idx = threadIdx.x; idx < P * Hc; idx += 256) {
        const int p = idx / Hc;
        const int h = idx - p * Hc;
        const size_t il = ((size_t)(b * Lc + p)) * Hc + h;
        const size_t ir = ((size_t)(b * Lc + p + 1)) * Hc + h;
        const float hl = h_old[il], hr = h_old[ir];
        const float cl = c_old[il], cr = c_old[ir];
        const float nhv = nh[((size_t)(b * P + p)) * Hc + h];
        const float ncv = nc[((size_t)(b * P + p)) * Hc + h];

        const float ch = (p < ksel) ? hl : ((p == ksel) ? nhv : hr);
        const float cc = (p < ksel) ? cl : ((p == ksel) ? ncv : cr);
        const float uh = is_last ? nhv : ch;
        const float uc = is_last ? ncv : cc;
        const float hn = done * uh + (1.0f - done) * hl;
        const float cn = done * uc + (1.0f - done) * cl;
        h_new[il] = hn;
        c_new[il] = cn;

        if (!is_last) {
            if (p == ksel) node_out[h] = nhv;           // sel_h == nh[argmax]
        } else if (p == 0) {
            node_out[h]        = hn;                    // h2[:,0,:]
            out_hf[b * Hc + h] = hn;
            out_cf[b * Hc + h] = cn;
        }
    }
    // pad row L-1 with zeros
    for (int h = threadIdx.x; h < Hc; h += 256) {
        const size_t ip = ((size_t)(b * Lc + (Lc - 1))) * Hc + h;
        h_new[ip] = 0.0f;
        c_new[ip] = 0.0f;
    }
}

// ---------------------------------------------------------------------------
// Host-side orchestration.
// ---------------------------------------------------------------------------
extern "C" void kernel_launch(void* const* d_in, const int* in_sizes, int n_in,
                              void* d_out, int out_size, void* d_ws, size_t ws_size,
                              hipStream_t stream) {
    (void)in_sizes; (void)n_in; (void)out_size; (void)ws_size;

    const float* inp        = (const float*)d_in[0];   // (B, L, D)
    const int*   length     = (const int*)  d_in[1];   // (B,)
    const float* W_word     = (const float*)d_in[2];   // (D, 2H)
    const float* b_word     = (const float*)d_in[3];   // (2H,)
    const float* W_comp     = (const float*)d_in[4];   // (2H, 5H)
    const float* b_comp     = (const float*)d_in[5];   // (5H,)
    const float* comp_query = (const float*)d_in[6];   // (H,)

    float* out      = (float*)d_out;
    float* out_hf   = out;                       // (B, H)
    float* out_cf   = out + Bc * Hc;             // (B, H)
    float* out_node = out + 2 * Bc * Hc;         // (B, 2L-1, H)

    // Workspace carve-out (256B aligned slabs).
    char*  ws  = (char*)d_ws;
    size_t off = 0;
    auto carve = [&](size_t bytes) -> void* {
        void* p = ws + off;
        off = (off + bytes + 255) & ~(size_t)255;
        return p;
    };
    bf16_t* A1    = (bf16_t*)carve((size_t)Bc * Lc * Dc * sizeof(bf16_t));   // leaf inputs, bf16
    bf16_t* WtW   = (bf16_t*)carve((size_t)NW * KW * sizeof(bf16_t));        // W_word^T bf16
    bf16_t* WtC   = (bf16_t*)carve((size_t)NC * KC * sizeof(bf16_t));        // W_comp^T bf16
    float*  hb[2] = { (float*)carve((size_t)Bc * Lc * Hc * sizeof(float)),
                      (float*)carve((size_t)Bc * Lc * Hc * sizeof(float)) };
    float*  cb[2] = { (float*)carve((size_t)Bc * Lc * Hc * sizeof(float)),
                      (float*)carve((size_t)Bc * Lc * Hc * sizeof(float)) };
    bf16_t* Acat  = (bf16_t*)carve((size_t)ROWS * KC * sizeof(bf16_t));
    float*  gates = (float*)carve((size_t)ROWS * NC * sizeof(float));
    float*  nhbuf = (float*)carve((size_t)ROWS * Hc * sizeof(float));
    float*  ncbuf = (float*)carve((size_t)ROWS * Hc * sizeof(float));
    float*  lgts  = (float*)carve((size_t)ROWS * sizeof(float));

    const size_t gemm_lds = 2 * (size_t)BUFB;    // 139,264 B double buffer

    // --- prep: bf16 conversions (deterministic per launch) ---
    {
        const int n = Bc * Lc * Dc;
        cvt_bf16_kernel<<<(n + 255) / 256, 256, 0, stream>>>(inp, A1, n);
    }
    {
        const int n = NW * KW;
        cvt_transpose_kernel<<<(n + 255) / 256, 256, 0, stream>>>(W_word, WtW, KW, NW);
    }
    {
        const int n = NC * KC;
        cvt_transpose_kernel<<<(n + 255) / 256, 256, 0, stream>>>(W_comp, WtC, KC, NC);
    }

    // --- leaf projection: hc = inp @ W_word + b_word, split into h0/c0 ---
    gemm_word_kernel<<<dim3(NW / BN, (Bc * Lc + BM - 1) / BM), 256, gemm_lds, stream>>>(
        A1, WtW, b_word, hb[0], cb[0]);

    // nodes[:, 0:L, :] = h0
    {
        const int n = Bc * Lc * Hc;
        copy_h0_kernel<<<(n + 255) / 256, 256, 0, stream>>>(hb[0], out_node);
    }

    // --- 31 dependent reduction steps ---
    int cur = 0;
    for (int i = 0; i < P; ++i) {
        {
            const int n = ROWS * KC;
            pack_kernel<<<(n + 255) / 256, 256, 0, stream>>>(hb[cur], Acat);
        }
        gemm_comp_kernel<<<dim3(NC / BN, (ROWS + BM - 1) / BM), 256, gemm_lds, stream>>>(
            Acat, WtC, b_comp, gates);
        cell_kernel<<<ROWS, 256, 0, stream>>>(gates, cb[cur], comp_query,
                                              nhbuf, ncbuf, lgts);
        update_kernel<<<Bc, 256, 0, stream>>>(hb[cur], cb[cur], nhbuf, ncbuf,
                                              lgts, length, hb[1 - cur], cb[1 - cur],
                                              out_node, out_hf, out_cf, i);
        cur ^= 1;
    }
}
